// MultiModalSNN_65790309040720
// MI455X (gfx1250) — matrix operand order repaired and checked by reading, not compiled, not run on hardware
//
#include <hip/hip_runtime.h>
#include <hip/hip_bf16.h>
#include <math.h>

// ---------------------------------------------------------------------------
// MultiModal SNN loss, fused for MI455X (gfx1250, wave32, WMMA, LDS-staged).
//
// N = 8192, D = 256. Two fused passes: Gram-tile (f16 WMMA, f32 acc) ->
// exp/mask/row-sum epilogue -> masked finite mean. Sim matrices (2x256 MB)
// are never materialized. B-operand tiles are staged through LDS once per
// block (double-buffered), not once per wave: 8x less L2 traffic, and the
// next tile's global loads overlap the WMMA chain.
// ---------------------------------------------------------------------------

typedef _Float16 h8   __attribute__((ext_vector_type(8)));
typedef _Float16 v16h __attribute__((ext_vector_type(16)));
typedef float    v8f  __attribute__((ext_vector_type(8)));

#define NTOT   8192
#define DDIM   256
#define BATCHN 2048
#define NMODS  4

#define NJT    512              // 16-col tiles across N
#define JSPLIT 4                // j-range split factor (parallelism)
#define JPB    (NJT / JSPLIT)   // j-tiles per block

#define LDSROW 264              // halves per padded LDS row (528 B = 132 dwords)
                                // -> ds_load_b128 lane banks = lane*4 (conflict-free)

union AB16 { v16h v; h8 h[2]; };

// --- zero acc[4] + num/den arrays (32832 floats) -----------------------------
__global__ void snn_init_ws(float* p, int n) {
    int i = blockIdx.x * blockDim.x + threadIdx.x;
    if (i < n) p[i] = 0.0f;
}

// --- mvec[i] = modality_mask[b][q] with i = q*BATCH + b (modality-major) -----
__global__ void snn_build_mask(const int* __restrict__ mm, float* __restrict__ mvec) {
    int i = blockIdx.x * blockDim.x + threadIdx.x;
    if (i < NTOT) {
        int b = i & (BATCHN - 1);
        int q = i >> 11;
        mvec[i] = (float)mm[b * NMODS + q];
    }
}

// --- row-normalize fp32 -> f16; one wave per row (8 elems/lane) --------------
__global__ __launch_bounds__(256)
void snn_normalize(const float* __restrict__ src, _Float16* __restrict__ dst) {
    int wave = (int)((blockIdx.x * blockDim.x + threadIdx.x) >> 5);
    int lane = threadIdx.x & 31;
    const float* row = src + (size_t)wave * DDIM;
    float v[8];
    float ss = 0.0f;
#pragma unroll
    for (int r = 0; r < 8; ++r) {
        v[r] = row[lane + 32 * r];
        ss += v[r] * v[r];
    }
#pragma unroll
    for (int m = 16; m >= 1; m >>= 1) ss += __shfl_xor(ss, m, 32);
    float rn = rsqrtf(ss);
    _Float16* drow = dst + (size_t)wave * DDIM;
#pragma unroll
    for (int r = 0; r < 8; ++r) drow[lane + 32 * r] = (_Float16)(v[r] * rn);
}

// --- fused Gram + SNN partial softmax sums ----------------------------------
// Block = 8 waves, each owning a 16-row output tile (128 rows/block).
// Block walks JPB column tiles; each tile staged once into LDS (dbl-buffered).
// mode: 0 = instance positives (i%2048), 1 = modality positives (i/2048).
__global__ __launch_bounds__(256)
void snn_gram(const _Float16* __restrict__ Z,
              const float* __restrict__ mvec,
              const float* __restrict__ log_tau,
              int mode,
              float* __restrict__ numArr,
              float* __restrict__ denArr) {
    __shared__ _Float16 lds[2][16 * LDSROW];   // 2 x 8448 B

    const int tid  = threadIdx.x;
    const int lane = tid & 31;
    const int wib  = tid >> 5;                 // wave in block (0..7)
    const int s    = blockIdx.x % JSPLIT;      // j-range slice
    const int ib   = blockIdx.x / JSPLIT;      // row-block (0..63)
    const int i0   = (ib * 8 + wib) * 16;
    const int half = lane >> 4;                // 0 or 1
    const int ln   = lane & 15;
    const int jt0  = s * JPB;

    const float inv_tau = __expf(-log_tau[0]); // 1/tau, scalar

    // Preload A operand: rows i0..i0+15, all 8 K-chunks -> 64 VGPRs.
    // 16-bit A 16x32 layout: lanes 0-15 hold K=[0..7]+[16..23],
    // lanes 16-31 hold K=[8..15]+[24..31] of row M = lane&15.
    v16h A[8];
    {
        const _Float16* arow = Z + (size_t)(i0 + ln) * DDIM;
#pragma unroll
        for (int kc = 0; kc < 8; ++kc) {
            int base = kc * 32 + half * 8;
            AB16 u;
            u.h[0] = *(const h8*)(arow + base);
            u.h[1] = *(const h8*)(arow + base + 16);
            A[kc] = u.v;
        }
    }

    float mi[8];
#pragma unroll
    for (int r = 0; r < 8; ++r) mi[r] = mvec[i0 + half * 8 + r];

    float num[8], den[8];
#pragma unroll
    for (int r = 0; r < 8; ++r) { num[r] = 0.0f; den[r] = 0.0f; }

    // Cooperative loader: 256 threads x 32 B = one 8 KB column tile.
    const int lrow = tid >> 4;   // 0..15 : row within tile
    const int lch  = tid & 15;   // 0..15 : 16-half (32 B) chunk within row
    {
        const _Float16* src = Z + (size_t)(jt0 * 16 + lrow) * DDIM + lch * 16;
        uint4 a = *(const uint4*)src;
        uint4 b = *(const uint4*)(src + 8);
        _Float16* d = &lds[0][lrow * LDSROW + lch * 16];
        *(uint4*)d = a;
        *(uint4*)(d + 8) = b;
    }

    for (int t = 0; t < JPB; ++t) {
        const int cur = t & 1;
        __syncthreads();   // lds[cur] writes done; lds[cur^1] readers done

        // Issue next tile's global loads before the math (latency overlap).
        uint4 pa{}, pb{};
        const bool more = (t + 1) < JPB;       // wave-uniform
        if (more) {
            const _Float16* src =
                Z + (size_t)((jt0 + t + 1) * 16 + lrow) * DDIM + lch * 16;
            pa = *(const uint4*)src;
            pb = *(const uint4*)(src + 8);
        }

        // Pull all 8 B chunks from LDS (bank-conflict-free), then 8 WMMAs.
        // B 32x16 layout: lane = half*16 + N, K=[8h..8h+7]+[16+8h..23+8h] per v16h.
        v8f c = {};
        {
            const _Float16* lb = &lds[cur][ln * LDSROW + half * 8];
            v16h B[8];
#pragma unroll
            for (int kc = 0; kc < 8; ++kc) {
                AB16 u;
                u.h[0] = *(const h8*)(lb + kc * 32);
                u.h[1] = *(const h8*)(lb + kc * 32 + 16);
                B[kc] = u.v;
            }
#pragma unroll
            for (int kc = 0; kc < 8; ++kc)
                c = __builtin_amdgcn_wmma_f32_16x16x32_f16(
                        false, A[kc], false, B[kc], (short)0, c, false, false);
        }

        // Park the prefetched tile into the other LDS buffer.
        if (more) {
            _Float16* d = &lds[cur ^ 1][lrow * LDSROW + lch * 16];
            *(uint4*)d = pa;
            *(uint4*)(d + 8) = pb;
        }

        // Epilogue: lane holds sim[i0 + half*8 + r][j] in c[r].
        const int j   = (jt0 + t) * 16 + ln;
        const float mj = mvec[j];
#pragma unroll
        for (int r = 0; r < 8; ++r) {
            int i = i0 + half * 8 + r;
            float w = mi[r] * mj;
            if (i == j) w = 0.0f;                          // exclude diagonal
            float e = __expf(c[r] * inv_tau) * w;
            den[r] += e;
            bool pos = mode ? ((i >> 11) == (j >> 11))     // same modality
                            : ((i & 2047) == (j & 2047));  // same instance
            num[r] += pos ? e : 0.0f;
        }
    }

    // Reduce across the 16 lanes (columns) of each half, then commit.
#pragma unroll
    for (int r = 0; r < 8; ++r) {
#pragma unroll
        for (int m = 1; m < 16; m <<= 1) {
            num[r] += __shfl_xor(num[r], m, 32);
            den[r] += __shfl_xor(den[r], m, 32);
        }
    }
    if (ln == 0) {   // lanes 0 and 16 -> rows i0+0..7 and i0+8..15
#pragma unroll
        for (int r = 0; r < 8; ++r) {
            int i = i0 + half * 8 + r;
            atomicAdd(&numArr[i], num[r]);
            atomicAdd(&denArr[i], den[r]);
        }
    }
}

// --- per-row loss + masked finite mean accumulation -------------------------
__global__ __launch_bounds__(256)
void snn_rowloss(const float* __restrict__ numI, const float* __restrict__ denI,
                 const float* __restrict__ numM, const float* __restrict__ denM,
                 const float* __restrict__ mvec, float* __restrict__ acc) {
    int idx = blockIdx.x * blockDim.x + threadIdx.x;   // 0..16383
    int mat = idx >> 13;                               // uniform per wave
    int i   = idx & (NTOT - 1);
    float n = mat ? numM[i] : numI[i];
    float d = mat ? denM[i] : denI[i];
    float ls = 0.0f, lc = 0.0f;
    if (mvec[i] > 0.0f && n > 0.0f && d > 0.0f) {
        float loss = __logf(d) - __logf(n);            // -log(num/den)
        if (loss == loss) { ls = loss; lc = 1.0f; }
    }
#pragma unroll
    for (int m = 16; m >= 1; m >>= 1) {
        ls += __shfl_xor(ls, m, 32);
        lc += __shfl_xor(lc, m, 32);
    }
    if ((threadIdx.x & 31) == 0) {
        atomicAdd(&acc[2 * mat], ls);
        atomicAdd(&acc[2 * mat + 1], lc);
    }
}

// --- finalize: two scalar means ---------------------------------------------
__global__ void snn_finalize(const float* __restrict__ acc, float* __restrict__ out) {
    if (threadIdx.x == 0) {
        out[0] = acc[0] / acc[1];
        out[1] = acc[2] / acc[3];
    }
}

extern "C" void kernel_launch(void* const* d_in, const int* in_sizes, int n_in,
                              void* d_out, int out_size, void* d_ws, size_t ws_size,
                              hipStream_t stream) {
    (void)in_sizes; (void)n_in; (void)out_size; (void)ws_size;

    const float* z_c    = (const float*)d_in[0];
    const float* z_t    = (const float*)d_in[1];
    const int*   mmask  = (const int*)d_in[2];
    const float* ltau_i = (const float*)d_in[3];
    const float* ltau_m = (const float*)d_in[4];
    float* out = (float*)d_out;

    // Workspace layout (~8.5 MB):
    //   [0B)      acc[4]: sum_i, cnt_i, sum_m, cnt_m
    //   [256B)    numI, denI, numM, denM: 4 x 8192 floats (128 KB)
    //   [+128KB)  mvec: 8192 floats (32 KB)
    //   [+32KB)   Zc: 8192x256 f16 (4 MB)
    //   [+4MB)    Zt: 8192x256 f16 (4 MB)
    char* ws = (char*)d_ws;
    float*    acc  = (float*)ws;
    float*    numI = (float*)(ws + 256);
    float*    denI = numI + NTOT;
    float*    numM = denI + NTOT;
    float*    denM = numM + NTOT;
    float*    mvec = denM + NTOT;
    _Float16* Zc   = (_Float16*)(mvec + NTOT);
    _Float16* Zt   = Zc + (size_t)NTOT * DDIM;

    // Zero acc + the four row arrays (contiguous from ws start): 64 + 4*8192.
    const int nz = 64 + 4 * NTOT;
    snn_init_ws<<<(nz + 255) / 256, 256, 0, stream>>>(acc, nz);
    snn_build_mask<<<NTOT / 256, 256, 0, stream>>>(mmask, mvec);

    // One wave per row -> 1024 blocks per matrix.
    snn_normalize<<<NTOT / 8, 256, 0, stream>>>(z_c, Zc);
    snn_normalize<<<NTOT / 8, 256, 0, stream>>>(z_t, Zt);

    // 64 row-blocks x JSPLIT j-slices = 256 blocks per pass; exact fit, no
    // divergence around the WMMAs (EXEC all-1s requirement).
    snn_gram<<<64 * JSPLIT, 256, 0, stream>>>(Zc, mvec, ltau_i, 0, numI, denI);
    snn_gram<<<64 * JSPLIT, 256, 0, stream>>>(Zt, mvec, ltau_m, 1, numM, denM);

    snn_rowloss<<<(2 * NTOT) / 256, 256, 0, stream>>>(numI, denI, numM, denM, mvec, acc);
    snn_finalize<<<1, 32, 0, stream>>>(acc, out);
}
